// CVAtten_15204184228561
// MI455X (gfx1250) — matrix operand order repaired and checked by reading, not compiled, not run on hardware
//
#include <hip/hip_runtime.h>

// ---------------- CDNA5 types ----------------
typedef _Float16 half_t;
typedef __attribute__((ext_vector_type(16))) _Float16 v16h;
typedef __attribute__((ext_vector_type(8)))  float    v8f;
typedef __attribute__((ext_vector_type(4)))  unsigned int v4u;

#define Bc   2
#define Cc   64
#define Hc   128
#define Wc   128
#define HWc  (Hc * Wc)
#define NPIX (Bc * HWc)
#define Dc   12
#define NKEY 24

// ---------------- WMMA helpers (16x16x32 f16 -> f32) ----------------
__device__ __forceinline__ v8f wmma_ff(v16h a, v16h b, v8f c) {
  return __builtin_amdgcn_wmma_f32_16x16x32_f16(false, a, false, b, (short)0, c,
                                                false, false);
}

// A fragment: 16x32 f16 tile from row-major LDS buffer (row stride rs in f16).
// Lane L holds row M=L%16; K = kh*16+{0..15}, kh = L>>4.
__device__ __forceinline__ v16h load_A_frag(const half_t* A, int rs, int kbase,
                                            int lane) {
  int row = lane & 15;
  int kh  = (lane >> 4) & 1;
  const half_t* p = A + row * rs + kbase + kh * 16;
  union { struct { v4u lo, hi; } u; v16h h; } cv;
  cv.u.lo = *(const v4u*)(p);        // K = kbase + kh*16 + 0..7
  cv.u.hi = *(const v4u*)(p + 8);    // K = kbase + kh*16 + 8..15
  return cv.h;
}

// B fragment: 32x16 f16 tile from TRANSPOSED LDS weights wT[n][k] (row = col n,
// Ktot f16 per row).  Lane L: col n = nbase + L%16, K = kbase + (L>>4)*16 + 0..15
// -> 16 contiguous f16 (32B), same contiguous K order as load_A_frag.
__device__ __forceinline__ v16h load_B_frag(const half_t* wT, int Ktot,
                                            int nbase, int kbase, int lane) {
  int col = nbase + (lane & 15);
  int k0  = kbase + ((lane >> 4) & 1) * 16;
  return *(const v16h*)(wT + col * Ktot + k0);
}

// C tile (16x16 f32): lane L col = L%16, rows M = r + 8*(L>>4).
__device__ __forceinline__ void store_C_frag(float* O, int rs, int nbase,
                                             v8f acc, int lane) {
  int col = nbase + (lane & 15);
  int mb  = ((lane >> 4) & 1) * 8;
#pragma unroll
  for (int r = 0; r < 8; ++r) O[(mb + r) * rs + col] = acc[r];
}

// OUT[16][ncols] f32 = A[16][64]f16 @ W^T  (K = 64)
__device__ __forceinline__ void gemm_A64(const half_t* A, int Ars,
                                         const half_t* wT, float* O, int Ors,
                                         int ncols, int lane) {
  v16h a0 = load_A_frag(A, Ars, 0, lane);
  v16h a1 = load_A_frag(A, Ars, 32, lane);
  for (int nb = 0; nb < ncols; nb += 16) {
    v8f acc = {};
    acc = wmma_ff(a0, load_B_frag(wT, 64, nb, 0, lane), acc);
    acc = wmma_ff(a1, load_B_frag(wT, 64, nb, 32, lane), acc);
    store_C_frag(O, Ors, nb, acc, lane);
  }
}

// LayerNorm over 64 ch for 16 pixels (lanes 0..15), f32 LDS -> f16 LDS.
__device__ __forceinline__ void layernorm16(const float* src, const float* g,
                                            const float* bb, half_t* dst,
                                            int lane, bool addQpe) {
  if (lane < 16) {
    const float* row = src + lane * 64;
    float s1 = 0.f, s2 = 0.f;
#pragma unroll 8
    for (int c = 0; c < 64; ++c) {
      float pe = (addQpe && ((c >> 1) & 1)) ? 1.f : 0.f;  // pos_embed(0): cos(0)=1
      float v  = row[c] + pe;
      s1 += v; s2 += v * v;
    }
    float mu  = s1 * (1.f / 64.f);
    float var = s2 * (1.f / 64.f) - mu * mu;
    float rs  = rsqrtf(var + 1e-5f);
#pragma unroll 8
    for (int c = 0; c < 64; ++c) {
      float pe = (addQpe && ((c >> 1) & 1)) ? 1.f : 0.f;
      float v  = row[c] + pe;
      dst[lane * 64 + c] = (half_t)((v - mu) * rs * g[c] + bb[c]);
    }
  }
}

// ---------------- fused per-pixel transformer kernel ----------------
// block = 128 threads (4 waves), wave owns 16 pixels, grid = NPIX/64 blocks.
__global__ __launch_bounds__(128, 1) void cvatten_main(
    const float* __restrict__ imgs, const float* __restrict__ pmat,
    const float* __restrict__ idm, const float* __restrict__ dint,
    const float* __restrict__ ln1g, const float* __restrict__ ln1b,
    const float* __restrict__ ln2g, const float* __restrict__ ln2b,
    const float* __restrict__ ln3g, const float* __restrict__ ln3b,
    const float* __restrict__ a1wv, const float* __restrict__ a1wo,
    const float* __restrict__ a1bo, const float* __restrict__ a2wq,
    const float* __restrict__ a2wk, const float* __restrict__ a2wv,
    const float* __restrict__ a2wo, const float* __restrict__ a2bo,
    const float* __restrict__ ffw1, const float* __restrict__ ffb1,
    const float* __restrict__ ffw2, const float* __restrict__ ffb2,
    float* __restrict__ out) {
  extern __shared__ char smem[];
  // ---- shared (block-wide) transposed f16 weights: 147456 B ----
  half_t* wv1T = (half_t*)smem;        // 64x64  -> [n][k], 4096 f16 each
  half_t* wo1T = wv1T + 4096;
  half_t* wq2T = wo1T + 4096;
  half_t* wk2T = wq2T + 4096;
  half_t* wv2T = wk2T + 4096;
  half_t* wo2T = wv2T + 4096;
  half_t* w1T  = wo2T + 4096;          // ff_w1 (64,512) -> [512][64]
  half_t* w2T  = w1T + 32768;          // ff_w2 (256,64) -> [64][256]
  char*   pw   = (char*)(w2T + 16384);

  const int wave = threadIdx.x >> 5;
  const int lane = threadIdx.x & 31;

  // ---- per-wave scratch: 27648 B each ----
  char*   ws   = pw + wave * 27648;
  half_t* tok  = (half_t*)ws;            // 16 x 256 f16 (A staging / FF mid)
  half_t* tokB = tok + 16 * 256;         // 16 x 64 f16  (A staging)
  float*  outb = (float*)(tokB + 16 * 64);  // 16 x 64 f32 GEMM out
  float*  obuf = outb + 16 * 64;            // 16 x 64 f32 attn accumulator
  float*  xbuf = obuf + 16 * 64;            // 16 x 64 f32 residual
  float*  qbuf = xbuf + 16 * 64;            // 16 x 64 f32 query proj
  float*  scal = qbuf + 16 * 64;            // 256 f32 scalars

  // ---- stage all weights transposed into LDS (f32 -> f16) ----
  {
    struct WDesc { const float* w; half_t* d; int K, N; };
    const WDesc wd[8] = {{a1wv, wv1T, 64, 64},  {a1wo, wo1T, 64, 64},
                         {a2wq, wq2T, 64, 64},  {a2wk, wk2T, 64, 64},
                         {a2wv, wv2T, 64, 64},  {a2wo, wo2T, 64, 64},
                         {ffw1, w1T, 64, 512},  {ffw2, w2T, 256, 64}};
#pragma unroll 1
    for (int t = 0; t < 8; ++t) {
      int KN = wd[t].K * wd[t].N;
      for (int i = threadIdx.x; i < KN; i += blockDim.x) {
        int k = i / wd[t].N, n = i % wd[t].N;
        wd[t].d[n * wd[t].K + k] = (half_t)wd[t].w[i];
      }
    }
  }
  __syncthreads();

  const int   pix0 = blockIdx.x * 64 + wave * 16;
  const int   p2   = lane >> 1;              // pixel sub-index for (pix, half) lanes
  const int   cb   = (lane & 1) * 32;        // channel base for (pix, half) lanes
  const int   pixL = pix0 + p2;
  const int   bL   = pixL >> 14;             // / HW
  const int   remL = pixL & (HWc - 1);

  __builtin_prefetch(imgs + (size_t)(bL * 3 * Cc) * HWc + remL, 0, 3);

  // ---- load query x0 = imgs[b,0,:,y,x] into xbuf ----
  {
    const float* qp = imgs + (size_t)(bL * 3 * Cc) * HWc + remL;
#pragma unroll 8
    for (int i = 0; i < 32; ++i)
      xbuf[p2 * 64 + cb + i] = qp[(size_t)(cb + i) * HWc];
  }

  // ---- attn1 (degenerate softmax): x += (LN1(x+qpe) @ wv1) @ wo1 + bo1 ----
  layernorm16(xbuf, ln1g, ln1b, tokB, lane, true);
  gemm_A64(tokB, 64, wv1T, outb, 64, 64, lane);
  for (int i = 0; i < 32; ++i)
    tokB[p2 * 64 + cb + i] = (half_t)outb[p2 * 64 + cb + i];
  gemm_A64(tokB, 64, wo1T, outb, 64, 64, lane);
  for (int i = 0; i < 32; ++i) {
    int c = cb + i;
    xbuf[p2 * 64 + c] += outb[p2 * 64 + c] + a1bo[c];
  }

  // ---- attn2 query: q = LN2(x) @ wq2 ----
  layernorm16(xbuf, ln2g, ln2b, tokB, lane, false);
  gemm_A64(tokB, 64, wq2T, qbuf, 64, 64, lane);

  // ---- online-softmax state ----
  if (lane < 16) {
    scal[64 + lane * 2 + 0] = -1e30f;  // m per head
    scal[64 + lane * 2 + 1] = -1e30f;
    scal[96 + lane * 2 + 0] = 0.f;     // s per head
    scal[96 + lane * 2 + 1] = 0.f;
  }
  for (int i = 0; i < 32; ++i) obuf[p2 * 64 + cb + i] = 0.f;

  const float depBase = idm[0];
  const float di      = dint[0];

  // ---- 24 warped key tokens: sample + pos-embed + WMMA K/V proj + softmax ----
  for (int k = 0; k < NKEY; ++k) {
    const int vsrc = k / Dc;
    const int dd   = k - vsrc * Dc;
    if (lane < 16) {
      int   pixel = pix0 + lane;
      int   b = pixel >> 14, rem = pixel & (HWc - 1);
      float x = (float)(rem & (Wc - 1)), y = (float)(rem >> 7);
      float dep = depBase + di * (float)dd;
      const float* P = pmat + ((b * 2 + vsrc) * 3 + 2) * 12;  // [b][v][LEVEL=2]
      float sx = P[0] * x + P[1] * y + P[2]  + P[3]  / dep;
      float sy = P[4] * x + P[5] * y + P[6]  + P[7]  / dep;
      float sz = P[8] * x + P[9] * y + P[10] + P[11] / dep;
      if (sz <= 1e-7f) { sx = 128.f; sy = 128.f; sz = 1.f; }
      float xs = sx / sz, ys = sy / sz;
      scal[lane * 4 + 0] = xs * (2.f / 127.f) - 1.f;  // gx
      scal[lane * 4 + 1] = ys * (2.f / 127.f) - 1.f;  // gy
      scal[lane * 4 + 2] = xs;
      scal[lane * 4 + 3] = ys;
    }
    // bilinear sample 64 ch + positional embedding -> tok (f16)
    {
      float gx = scal[p2 * 4 + 0], gy = scal[p2 * 4 + 1];
      float xs = scal[p2 * 4 + 2], ys = scal[p2 * 4 + 3];
      float x0f = floorf(xs), y0f = floorf(ys);
      float fx = xs - x0f, fy = ys - y0f;
      float vx0 = (x0f >= 0.f && x0f <= 127.f) ? 1.f : 0.f;
      float vx1 = (x0f + 1.f >= 0.f && x0f + 1.f <= 127.f) ? 1.f : 0.f;
      float vy0 = (y0f >= 0.f && y0f <= 127.f) ? 1.f : 0.f;
      float vy1 = (y0f + 1.f >= 0.f && y0f + 1.f <= 127.f) ? 1.f : 0.f;
      float w00 = (1.f - fx) * (1.f - fy) * vx0 * vy0;
      float w01 = fx * (1.f - fy) * vx1 * vy0;
      float w10 = (1.f - fx) * fy * vx0 * vy1;
      float w11 = fx * fy * vx1 * vy1;
      int cx0 = (int)fminf(fmaxf(x0f, 0.f), 127.f);
      int cx1 = (int)fminf(fmaxf(x0f + 1.f, 0.f), 127.f);
      int cy0 = (int)fminf(fmaxf(y0f, 0.f), 127.f);
      int cy1 = (int)fminf(fmaxf(y0f + 1.f, 0.f), 127.f);
      int o00 = cy0 * Wc + cx0, o01 = cy0 * Wc + cx1;
      int o10 = cy1 * Wc + cx0, o11 = cy1 * Wc + cx1;
      const float* fb =
          imgs + (size_t)((bL * 3 + 1 + vsrc) * Cc) * HWc;
#pragma unroll 4
      for (int i = 0; i < 32; ++i) {
        int c = cb + i;
        const float* fc = fb + (size_t)c * HWc;
        float val = w00 * fc[o00] + w01 * fc[o01] + w10 * fc[o10] +
                    w11 * fc[o11];
        int   f = c >> 2;
        float coord = (c & 1) ? gy : gx;
        float ang   = coord * (float)(1 << f);
        float pe    = ((c >> 1) & 1) ? __cosf(ang) : __sinf(ang);
        tok[p2 * 256 + c] = (half_t)(val + pe);
      }
    }
    // Kp = token @ wk2 ; logits + online softmax
    gemm_A64(tok, 256, wk2T, outb, 64, 64, lane);
    if (lane < 16) {
#pragma unroll
      for (int h = 0; h < 2; ++h) {
        float l = 0.f;
#pragma unroll 8
        for (int j = 0; j < 32; ++j)
          l += qbuf[lane * 64 + h * 32 + j] * outb[lane * 64 + h * 32 + j];
        l *= 0.17677669529663687f;  // 1/sqrt(32)
        float mo = scal[64 + lane * 2 + h];
        float mn = fmaxf(mo, l);
        float sc = __expf(mo - mn);
        float wg = __expf(l - mn);
        scal[96  + lane * 2 + h] = scal[96 + lane * 2 + h] * sc + wg;
        scal[64  + lane * 2 + h] = mn;
        scal[128 + lane * 2 + h] = sc;
        scal[160 + lane * 2 + h] = wg;
      }
    }
    // Vp = token @ wv2 ; o = o*sc + wg*Vp  (lane's 32 ch all in head = lane&1)
    gemm_A64(tok, 256, wv2T, outb, 64, 64, lane);
    {
      int   h  = lane & 1;
      float sc = scal[128 + p2 * 2 + h];
      float wg = scal[160 + p2 * 2 + h];
#pragma unroll 8
      for (int i = 0; i < 32; ++i) {
        int c = cb + i;
        obuf[p2 * 64 + c] = obuf[p2 * 64 + c] * sc + wg * outb[p2 * 64 + c];
      }
    }
  }

  // ---- attn2 output: x += (o/s) @ wo2 + bo2 ----
  {
    int   h    = lane & 1;
    float sinv = 1.f / scal[96 + p2 * 2 + h];
    for (int i = 0; i < 32; ++i)
      tokB[p2 * 64 + cb + i] = (half_t)(obuf[p2 * 64 + cb + i] * sinv);
  }
  gemm_A64(tokB, 64, wo2T, outb, 64, 64, lane);
  for (int i = 0; i < 32; ++i) {
    int c = cb + i;
    xbuf[p2 * 64 + c] += outb[p2 * 64 + c] + a2bo[c];
  }

  // ---- FF: hh = LN3(x)@ff_w1 + b1 ; mid = a*gelu(gate) ; y = mid@ff_w2 ----
  layernorm16(xbuf, ln3g, ln3b, tokB, lane, false);
  {
    v16h a0 = load_A_frag(tokB, 64, 0, lane);
    v16h a1 = load_A_frag(tokB, 64, 32, lane);
    int  col16 = lane & 15;
    int  mb    = ((lane >> 4) & 1) * 8;
#pragma unroll 1
    for (int j = 0; j < 16; ++j) {
      v8f aa = {}, gg = {};
      aa = wmma_ff(a0, load_B_frag(w1T, 64, j * 16, 0, lane), aa);
      aa = wmma_ff(a1, load_B_frag(w1T, 64, j * 16, 32, lane), aa);
      gg = wmma_ff(a0, load_B_frag(w1T, 64, 256 + j * 16, 0, lane), gg);
      gg = wmma_ff(a1, load_B_frag(w1T, 64, 256 + j * 16, 32, lane), gg);
      float ba = ffb1[j * 16 + col16];
      float bg = ffb1[256 + j * 16 + col16];
      int   cl = j * 16 + col16;
#pragma unroll
      for (int r = 0; r < 8; ++r) {
        float av = aa[r] + ba;
        float gv = gg[r] + bg;
        float t  = gv + 0.044715f * gv * gv * gv;
        float ge = 0.5f * gv * (1.f + tanhf(0.7978845608f * t));
        tok[(mb + r) * 256 + cl] = (half_t)(av * ge);
      }
    }
  }
  // mid (16x256) @ ff_w2 (K=256)
  for (int nb = 0; nb < 64; nb += 16) {
    v8f acc = {};
#pragma unroll
    for (int kb = 0; kb < 256; kb += 32)
      acc = wmma_ff(load_A_frag(tok, 256, kb, lane),
                    load_B_frag(w2T, 256, nb, kb, lane), acc);
    store_C_frag(outb, 64, nb, acc, lane);
  }

  // ---- write out[b,0,c,y,x] = x + ff_out + ff_b2 ----
  {
    float* op = out + (size_t)(bL * 3 * Cc) * HWc + remL;
#pragma unroll 8
    for (int i = 0; i < 32; ++i) {
      int c = cb + i;
      op[(size_t)c * HWc] = xbuf[p2 * 64 + c] + outb[p2 * 64 + c] + ffb2[c];
    }
  }
}

// out[:,1:3] = imgs[:,1:3]  (identical flat offsets since V==3)
__global__ __launch_bounds__(256) void copy_views(const float* __restrict__ imgs,
                                                  float* __restrict__ out) {
  const size_t chunk = (size_t)2 * Cc * HWc;      // per-batch views 1..2
  size_t i = (size_t)blockIdx.x * blockDim.x + threadIdx.x;
  if (i >= (2 * chunk) / 4) return;
  size_t e = i * 4;
  size_t b = e / chunk, r = e - b * chunk;
  size_t off = b * (size_t)3 * Cc * HWc + (size_t)Cc * HWc + r;
  *(float4*)(out + off) = *(const float4*)(imgs + off);
}

extern "C" void kernel_launch(void* const* d_in, const int* in_sizes, int n_in,
                              void* d_out, int out_size, void* d_ws,
                              size_t ws_size, hipStream_t stream) {
  const float* imgs = (const float*)d_in[0];
  const float* pmat = (const float*)d_in[1];
  const float* idm  = (const float*)d_in[2];
  const float* dint = (const float*)d_in[3];
  // d_in[4] = m (unused)
  const float* ln1g = (const float*)d_in[5];
  const float* ln1b = (const float*)d_in[6];
  const float* ln2g = (const float*)d_in[7];
  const float* ln2b = (const float*)d_in[8];
  const float* ln3g = (const float*)d_in[9];
  const float* ln3b = (const float*)d_in[10];
  // 11 a1_wq, 12 a1_wk unused (degenerate single-token self-attention)
  const float* a1wv = (const float*)d_in[13];
  const float* a1wo = (const float*)d_in[14];
  const float* a1bo = (const float*)d_in[15];
  const float* a2wq = (const float*)d_in[16];
  const float* a2wk = (const float*)d_in[17];
  const float* a2wv = (const float*)d_in[18];
  const float* a2wo = (const float*)d_in[19];
  const float* a2bo = (const float*)d_in[20];
  const float* ffw1 = (const float*)d_in[21];
  const float* ffb1 = (const float*)d_in[22];
  const float* ffw2 = (const float*)d_in[23];
  const float* ffb2 = (const float*)d_in[24];
  float* out = (float*)d_out;

  const size_t ldsBytes = 147456 + 4 * 27648;  // 258048 B (<= 320KB WGP LDS)
  cvatten_main<<<NPIX / 64, 128, ldsBytes, stream>>>(
      imgs, pmat, idm, dint, ln1g, ln1b, ln2g, ln2b, ln3g, ln3b, a1wv, a1wo,
      a1bo, a2wq, a2wk, a2wv, a2wo, a2bo, ffw1, ffb1, ffw2, ffb2, out);

  const size_t copyElems = (size_t)2 * 2 * Cc * HWc / 4;  // float4 units
  copy_views<<<(int)((copyElems + 255) / 256), 256, 0, stream>>>(imgs, out);
}
